// RWKV7TimeMixForTabICL_43628277793213
// MI455X (gfx1250) — matrix operand order repaired and checked by reading, not compiled, not run on hardware
//
#include <hip/hip_runtime.h>
#include <hip/hip_bf16.h>
#include <math.h>

// ---------------------------------------------------------------------------
// RWKV7 time-mix for MI455X (gfx1250): bf16 WMMA GEMMs + chunked scan.
// ---------------------------------------------------------------------------

typedef __attribute__((ext_vector_type(16))) __bf16 v16bf;
typedef __attribute__((ext_vector_type(8)))  __bf16 v8bf;
typedef __attribute__((ext_vector_type(8)))  float  v8f;
typedef int v4i_vs __attribute__((vector_size(16)));   // matches builtin proto

#define Bsz   4
#define Nseq  4096
#define Dm    1024
#define Hh    16
#define Mrows (Bsz * Nseq)        // 16384
#define GN_EPS 0.00064f

#define BM 64
#define BN 128
#define BK 32
#define NKT (Dm / BK)             // 32
#define LDA (BK + 8)              // padded LDS row (80B, 16B-aligned rows)

#define CHUNKS 16
#define CLEN   (Nseq / CHUNKS)    // 256

#if defined(__HIP_DEVICE_COMPILE__) && defined(__gfx1250__) && \
    __has_builtin(__builtin_amdgcn_global_load_async_to_lds_b128) && \
    __has_builtin(__builtin_amdgcn_s_wait_asynccnt)
#define HAS_ASYNC 1
#else
#define HAS_ASYNC 0
#endif

#if HAS_ASYNC
// global (AS1) -> LDS (AS3) 16-byte async copy, tracked by ASYNCcnt.
__device__ __forceinline__ void async_copy16(const void* g, void* l) {
  __builtin_amdgcn_global_load_async_to_lds_b128(
      (__attribute__((address_space(1))) v4i_vs*)(g),
      (__attribute__((address_space(3))) v4i_vs*)(l),
      0, 0);
}
#endif

__device__ __forceinline__ v16bf concat8(v8bf lo, v8bf hi) {
  return __builtin_shufflevector(lo, hi, 0, 1, 2, 3, 4, 5, 6, 7,
                                         8, 9, 10, 11, 12, 13, 14, 15);
}

__device__ __forceinline__ float sigm(float y) {
  return 1.0f / (1.0f + __expf(-y));
}

// --------------------------- weight f32 -> bf16 ----------------------------
__global__ __launch_bounds__(256) void convert_w_kernel(
    const float* __restrict__ src, __bf16* __restrict__ dst, int n) {
  int i = (blockIdx.x * 256 + threadIdx.x) * 4;
  if (i < n) {
    float4 v = *reinterpret_cast<const float4*>(src + i);
    dst[i + 0] = (__bf16)v.x;
    dst[i + 1] = (__bf16)v.y;
    dst[i + 2] = (__bf16)v.z;
    dst[i + 3] = (__bf16)v.w;
  }
}

// ------------------------------- WMMA GEMM ---------------------------------
// y[m,n] = sum_k A[m,k] * W[n,k]   (torch Linear: z @ W^T)
// Block tile 64x128, 8 waves as 2(m) x 4(n); each wave owns a 32x32 output
// (2x2 WMMA accumulators -> 4 v_wmma per k-step, 2x A/B fragment reuse).
// amode 0: A = x + (prev - x) * mix   (f32 sources, bf16 into LDS)
// amode 1: A = Abf (bf16, row-major)  -- async global->LDS when available
// act: 0 sigmoid, 1 sigmoid(-y) [= exp(-softplus)], 2 tanh, 3 identity,
//      4 u = sigmoid(y)*auxk*auxv, 5 sigmoid, 6 y + bias (final projection)
__global__ __launch_bounds__(256) void wmma_gemm_kernel(
    const float* __restrict__ x, const float* __restrict__ mix,
    const __bf16* __restrict__ Wb, const __bf16* __restrict__ Abf,
    float* outp, const float* auxk, const float* auxv,
    const float* __restrict__ bias, int act, int amode) {
  __shared__ __align__(16) __bf16 As[2][BM][LDA];
  __shared__ __align__(16) __bf16 Bs[2][BN][LDA];

  const int tid  = threadIdx.x;
  const int m0   = blockIdx.x * BM;
  const int n0   = blockIdx.y * BN;
  const int lrow = tid >> 2;        // 0..63  (A loader row)
  const int lk8  = (tid & 3) * 8;   // 0,8,16,24 (A loader k chunk)
  const int brow = tid >> 1;        // 0..127 (B loader row)
  const int bk16 = (tid & 1) * 16;  // 0,16   (B loader k chunk)

  auto loadA = [&](int k0, int buf) {
    if (amode == 0) {
      const int    gm = m0 + lrow;
      const size_t gb = (size_t)gm * Dm + (size_t)(k0 + lk8);
      float4 xv0 = *reinterpret_cast<const float4*>(x + gb);
      float4 xv1 = *reinterpret_cast<const float4*>(x + gb + 4);
      float4 pv0, pv1;
      if ((gm & (Nseq - 1)) == 0) {   // first token of its batch: prev = 0
        pv0 = make_float4(0.f, 0.f, 0.f, 0.f);
        pv1 = pv0;
      } else {
        pv0 = *reinterpret_cast<const float4*>(x + gb - Dm);
        pv1 = *reinterpret_cast<const float4*>(x + gb - Dm + 4);
      }
      float4 mv0 = *reinterpret_cast<const float4*>(mix + k0 + lk8);
      float4 mv1 = *reinterpret_cast<const float4*>(mix + k0 + lk8 + 4);
      v8bf z;
      z[0] = (__bf16)(xv0.x + (pv0.x - xv0.x) * mv0.x);
      z[1] = (__bf16)(xv0.y + (pv0.y - xv0.y) * mv0.y);
      z[2] = (__bf16)(xv0.z + (pv0.z - xv0.z) * mv0.z);
      z[3] = (__bf16)(xv0.w + (pv0.w - xv0.w) * mv0.w);
      z[4] = (__bf16)(xv1.x + (pv1.x - xv1.x) * mv1.x);
      z[5] = (__bf16)(xv1.y + (pv1.y - xv1.y) * mv1.y);
      z[6] = (__bf16)(xv1.z + (pv1.z - xv1.z) * mv1.z);
      z[7] = (__bf16)(xv1.w + (pv1.w - xv1.w) * mv1.w);
      *reinterpret_cast<v8bf*>(&As[buf][lrow][lk8]) = z;
    } else {
      const __bf16* gp = Abf + (size_t)(m0 + lrow) * Dm + (size_t)(k0 + lk8);
#if HAS_ASYNC
      async_copy16(gp, &As[buf][lrow][lk8]);
#else
      *reinterpret_cast<v8bf*>(&As[buf][lrow][lk8]) =
          *reinterpret_cast<const v8bf*>(gp);
#endif
    }
  };

  auto loadB = [&](int k0, int buf) {
    const __bf16* gp = Wb + (size_t)(n0 + brow) * Dm + (size_t)(k0 + bk16);
#if HAS_ASYNC
    async_copy16(gp, &Bs[buf][brow][bk16]);
    async_copy16(gp + 8, &Bs[buf][brow][bk16 + 8]);
#else
    *reinterpret_cast<v8bf*>(&Bs[buf][brow][bk16]) =
        *reinterpret_cast<const v8bf*>(gp);
    *reinterpret_cast<v8bf*>(&Bs[buf][brow][bk16 + 8]) =
        *reinterpret_cast<const v8bf*>(gp + 8);
#endif
  };

  const int wid  = tid >> 5;
  const int lane = tid & 31;
  const int wm   = wid & 1;        // wave row block (2 x 32 rows)
  const int wn   = wid >> 1;       // wave col block (4 x 32 cols)
  const int ln   = lane & 15;
  const int lh   = lane >> 4;      // K-half selector

  v8f acc00 = {}, acc01 = {}, acc10 = {}, acc11 = {};

  const int am0 = wm * 32 + ln;    // A rows for frag 0 / frag 1 (+16)
  const int bn0 = wn * 32 + ln;    // B cols for frag 0 / frag 1 (+16)
  const int kh  = lh * 8;          // A: K = {kh..kh+7, kh+16..kh+23}
  const int ks  = lh * 8;          // B: K = {ks..ks+7, ks+16..ks+23} (same as A)

  auto compute = [&](int buf) {
    v16bf a0 = concat8(*reinterpret_cast<const v8bf*>(&As[buf][am0][kh]),
                       *reinterpret_cast<const v8bf*>(&As[buf][am0][kh + 16]));
    v16bf a1 = concat8(*reinterpret_cast<const v8bf*>(&As[buf][am0 + 16][kh]),
                       *reinterpret_cast<const v8bf*>(&As[buf][am0 + 16][kh + 16]));
    v16bf b0 = concat8(*reinterpret_cast<const v8bf*>(&Bs[buf][bn0][ks]),
                       *reinterpret_cast<const v8bf*>(&Bs[buf][bn0][ks + 16]));
    v16bf b1 = concat8(*reinterpret_cast<const v8bf*>(&Bs[buf][bn0 + 16][ks]),
                       *reinterpret_cast<const v8bf*>(&Bs[buf][bn0 + 16][ks + 16]));
    acc00 = __builtin_amdgcn_wmma_f32_16x16x32_bf16(false, a0, false, b0,
                                                    (short)0, acc00, false, false);
    acc01 = __builtin_amdgcn_wmma_f32_16x16x32_bf16(false, a0, false, b1,
                                                    (short)0, acc01, false, false);
    acc10 = __builtin_amdgcn_wmma_f32_16x16x32_bf16(false, a1, false, b0,
                                                    (short)0, acc10, false, false);
    acc11 = __builtin_amdgcn_wmma_f32_16x16x32_bf16(false, a1, false, b1,
                                                    (short)0, acc11, false, false);
  };

  // Software pipeline: steady-state loop with unconditional prefetch,
  // final k-step peeled.  unroll 2 -> buffer index constant-folds.
  loadA(0, 0);
  loadB(0, 0);
  int buf = 0;
#pragma unroll 2
  for (int kt = 0; kt < NKT - 1; ++kt) {
#if HAS_ASYNC
    __builtin_amdgcn_s_wait_asynccnt(0);
#endif
    __syncthreads();
    loadA((kt + 1) * BK, buf ^ 1);
    loadB((kt + 1) * BK, buf ^ 1);
    compute(buf);
    __syncthreads();
    buf ^= 1;
  }
#if HAS_ASYNC
  __builtin_amdgcn_s_wait_asynccnt(0);
#endif
  __syncthreads();
  compute(buf);

  // Epilogue. C/D layout: m = base + lh*8 + vgpr, n = base + (lane&15).
#pragma unroll
  for (int ai = 0; ai < 2; ++ai) {
#pragma unroll
    for (int bi = 0; bi < 2; ++bi) {
      v8f a = ai ? (bi ? acc11 : acc10) : (bi ? acc01 : acc00);
      const int gmb = m0 + wm * 32 + ai * 16 + lh * 8;
      const int gn  = n0 + wn * 32 + bi * 16 + ln;
#pragma unroll
      for (int rr = 0; rr < 8; ++rr) {
        const size_t oi = (size_t)(gmb + rr) * Dm + gn;
        const float y = a[rr];
        float o;
        if (act == 0 || act == 5)      o = sigm(y);                       // r, g
        else if (act == 1)             o = 1.0f / (1.0f + __expf(y));     // decay
        else if (act == 2)             o = tanhf(y);                      // k
        else if (act == 3)             o = y;                             // v
        else if (act == 4)             o = sigm(y) * auxk[oi] * auxv[oi]; // u
        else                           o = y + bias[gn];                  // out
        outp[oi] = o;
      }
    }
  }
}

// ------------------------ chunked linear-scan (3 pass) ---------------------
__global__ __launch_bounds__(256) void scan_partA_kernel(
    const float* __restrict__ dec, const float* __restrict__ u,
    float2* __restrict__ part) {
  const int lin = blockIdx.x * 256 + threadIdx.x;   // 65536 threads
  const int c  = lin & (Dm - 1);
  const int b  = (lin >> 10) & (Bsz - 1);
  const int ck = lin >> 12;
  size_t base = ((size_t)b * Nseq + (size_t)ck * CLEN) * Dm + c;
  float P = 1.0f, S = 0.0f;
  for (int t = 0; t < CLEN; ++t) {
    const float d  = dec[base + (size_t)t * Dm];
    const float uu = u[base + (size_t)t * Dm];
    S = S * d + uu;
    P *= d;
  }
  part[(size_t)(b * Dm + c) * CHUNKS + ck] = make_float2(P, S);
}

__global__ __launch_bounds__(256) void scan_partB_kernel(
    const float2* __restrict__ part, float* __restrict__ carry) {
  const int lin = blockIdx.x * 256 + threadIdx.x;   // 4096 channels
  const size_t o = (size_t)lin * CHUNKS;
  float s = 0.0f;
  for (int j = 0; j < CHUNKS; ++j) {
    carry[o + j] = s;                // state entering chunk j
    float2 ps = part[o + j];
    s = s * ps.x + ps.y;
  }
}

// NOTE: `mixed` may alias `dec` (same-index read-before-write, single thread).
__global__ __launch_bounds__(256) void scan_partC_kernel(
    const float* __restrict__ r, const float* dec, const float* __restrict__ u,
    const float* __restrict__ carry, float* mixed) {
  const int lin = blockIdx.x * 256 + threadIdx.x;
  const int c  = lin & (Dm - 1);
  const int b  = (lin >> 10) & (Bsz - 1);
  const int ck = lin >> 12;
  size_t base = ((size_t)b * Nseq + (size_t)ck * CLEN) * Dm + c;
  float state = carry[(size_t)(b * Dm + c) * CHUNKS + ck];
  for (int t = 0; t < CLEN; ++t) {
    const size_t idx = base + (size_t)t * Dm;
    const float d  = dec[idx];
    const float uu = u[idx];
    const float rr = r[idx];
    state = state * d + uu;
    mixed[idx] = rr * state;
  }
}

// -------------------- GroupNorm(H=16) + gate -> bf16 A ---------------------
__global__ __launch_bounds__(256) void gnorm_gate_kernel(
    const float* __restrict__ mixed, const float* __restrict__ g,
    const float* __restrict__ gw, const float* __restrict__ gb,
    __bf16* __restrict__ hb) {
  const int wid  = threadIdx.x >> 5;
  const int lane = threadIdx.x & 31;
  const size_t grp = (size_t)blockIdx.x * 8 + wid;  // B*N*H groups
  const size_t row = grp >> 4;                      // token index in [0, M)
  const int    h   = (int)(grp & 15);
  const size_t base = row * Dm + (size_t)h * 64 + lane * 2;
  float2 v = *reinterpret_cast<const float2*>(mixed + base);
  float s  = v.x + v.y;
  float ss = v.x * v.x + v.y * v.y;
  for (int m = 16; m > 0; m >>= 1) {
    s  += __shfl_xor(s, m, 32);
    ss += __shfl_xor(ss, m, 32);
  }
  const float mu  = s * (1.0f / 64.0f);
  const float var = ss * (1.0f / 64.0f) - mu * mu;
  const float inv = rsqrtf(var + GN_EPS);
  const int ch = h * 64 + lane * 2;
  const float y0 = (v.x - mu) * inv * gw[ch] + gb[ch];
  const float y1 = (v.y - mu) * inv * gw[ch + 1] + gb[ch + 1];
  float2 gg = *reinterpret_cast<const float2*>(g + base);
  hb[base]     = (__bf16)(y0 * gg.x);
  hb[base + 1] = (__bf16)(y1 * gg.y);
}

// ------------------------------ host launcher ------------------------------
extern "C" void kernel_launch(void* const* d_in, const int* in_sizes, int n_in,
                              void* d_out, int out_size, void* d_ws,
                              size_t ws_size, hipStream_t stream) {
  const float* x    = (const float*)d_in[0];
  const float* mx_r = (const float*)d_in[1];
  const float* mx_w = (const float*)d_in[2];
  const float* mx_k = (const float*)d_in[3];
  const float* mx_v = (const float*)d_in[4];
  const float* mx_a = (const float*)d_in[5];
  const float* mx_g = (const float*)d_in[6];
  // W_r, W_w, W_k, W_v, W_a, W_g, W_out = d_in[7..13]
  const float* b_out = (const float*)d_in[14];
  const float* gn_w  = (const float*)d_in[15];
  const float* gn_b  = (const float*)d_in[16];

  // Workspace layout (bytes):
  //   [0, 14MB)        : 7 bf16 weight matrices (2MB each)
  //   +0*64MB          : bufR   (f32, r)
  //   +1*64MB          : bufW   (f32, decay -> overwritten with mixed)
  //   +2*64MB          : bufKU  (f32, k -> u; later reused as bf16 h-matrix)
  //   +3*64MB          : bufVG  (f32, v -> g)
  //   +4*64MB          : scan partials (512KB) + carries (256KB)
  const size_t WELEM = (size_t)Dm * Dm;                 // 1M elements
  const size_t WBsz  = 7 * WELEM * sizeof(__bf16);      // 14,680,064
  const size_t FB    = (size_t)Mrows * Dm * sizeof(float); // 64MB
  char* ws = (char*)d_ws;
  __bf16* wb    = (__bf16*)ws;
  float*  bufR  = (float*)(ws + WBsz);
  float*  bufW  = (float*)(ws + WBsz + 1 * FB);
  float*  bufKU = (float*)(ws + WBsz + 2 * FB);
  float*  bufVG = (float*)(ws + WBsz + 3 * FB);
  float2* part  = (float2*)(ws + WBsz + 4 * FB);
  float*  carry = (float*)((char*)part + (size_t)Bsz * Dm * CHUNKS * sizeof(float2));
  __bf16* hbuf  = (__bf16*)bufKU;   // reuse u region for bf16 gated/normed A

  // 1) weights -> bf16 (order: r, w, k, v, a, g, out)
  for (int i = 0; i < 7; ++i) {
    convert_w_kernel<<<(int)(WELEM / (256 * 4)), 256, 0, stream>>>(
        (const float*)d_in[7 + i], wb + (size_t)i * WELEM, (int)WELEM);
  }

  dim3 gg(Mrows / BM, Dm / BN);   // 256 x 8
  dim3 bb(256);

  // 2) six projection GEMMs (buffer-reuse-safe order)
  wmma_gemm_kernel<<<gg, bb, 0, stream>>>(x, mx_k, wb + 2 * WELEM, nullptr,
                                          bufKU, nullptr, nullptr, nullptr, 2, 0); // k=tanh
  wmma_gemm_kernel<<<gg, bb, 0, stream>>>(x, mx_v, wb + 3 * WELEM, nullptr,
                                          bufVG, nullptr, nullptr, nullptr, 3, 0); // v
  wmma_gemm_kernel<<<gg, bb, 0, stream>>>(x, mx_a, wb + 4 * WELEM, nullptr,
                                          bufKU, bufKU, bufVG, nullptr, 4, 0);     // u=σ(a)kv
  wmma_gemm_kernel<<<gg, bb, 0, stream>>>(x, mx_g, wb + 5 * WELEM, nullptr,
                                          bufVG, nullptr, nullptr, nullptr, 5, 0); // g=σ
  wmma_gemm_kernel<<<gg, bb, 0, stream>>>(x, mx_r, wb + 0 * WELEM, nullptr,
                                          bufR, nullptr, nullptr, nullptr, 0, 0);  // r=σ
  wmma_gemm_kernel<<<gg, bb, 0, stream>>>(x, mx_w, wb + 1 * WELEM, nullptr,
                                          bufW, nullptr, nullptr, nullptr, 1, 0);  // decay

  // 3) chunked scan: s_t = s_{t-1}*d_t + u_t ; mixed = r*s
  scan_partA_kernel<<<256, 256, 0, stream>>>(bufW, bufKU, part);
  scan_partB_kernel<<<16, 256, 0, stream>>>(part, carry);
  scan_partC_kernel<<<256, 256, 0, stream>>>(bufR, bufW, bufKU, carry, bufW);

  // 4) GroupNorm + gate -> bf16 A for output projection
  gnorm_gate_kernel<<<(Mrows * Hh) / 8, 256, 0, stream>>>(bufW, bufVG, gn_w,
                                                          gn_b, hbuf);

  // 5) final GEMM: out = h @ W_out^T + b_out
  wmma_gemm_kernel<<<gg, bb, 0, stream>>>(x, mx_r, wb + 6 * WELEM, hbuf,
                                          (float*)d_out, nullptr, nullptr,
                                          b_out, 6, 1);
}
